// HierarchicalFeatureEncoder_48352741818740
// MI455X (gfx1250) — matrix operand order repaired and checked by reading, not compiled, not run on hardware
//
#include <hip/hip_runtime.h>
#include <hip/hip_bf16.h>
#include <stdint.h>

typedef __attribute__((ext_vector_type(16))) _Float16 v16h;
typedef __attribute__((ext_vector_type(8)))  float    v8f;

#define WAVES 2   // waves per block; each wave owns a 16-row tile

// ---------------------------------------------------------------------------
// Prologue: pack an arbitrary (K x N) f32 matrix, element (k,n) = W[k*rs+n*cs],
// into WMMA B-fragment layout (f16).  Fragment f = kt*NT + nt holds the
// 32x16 tile; per lane the 8 dwords are stored contiguously:
//   out[(f*32 + lane)*8 + v] = pack( B[kt*32 + (lane>>4)*8 + 2v + (v>=4)*8    , nt*16 + (lane&15)],
//                                    B[kt*32 + (lane>>4)*8 + 2v + (v>=4)*8 + 1, ...               ] )
// which matches the 16-bit 32x16 B layout in CDNA5 ISA.
// ---------------------------------------------------------------------------
__global__ void pack_b_kernel(const float* __restrict__ W, int rs, int cs,
                              int KT, int NT, uint32_t* __restrict__ out) {
  int t = blockIdx.x * blockDim.x + threadIdx.x;
  int total = KT * NT * 256;
  if (t >= total) return;
  int v    = t & 7;
  int lane = (t >> 3) & 31;
  int f    = t >> 8;
  int nt = f % NT;
  int kt = f / NT;
  int n = nt * 16 + (lane & 15);
  int k = kt * 32 + ((lane >> 4) << 3) + 2 * v + ((v >> 2) << 3);
  union { _Float16 h[2]; uint32_t u; } pk;
  pk.h[0] = (_Float16)W[(size_t)k * rs + (size_t)n * cs];
  pk.h[1] = (_Float16)W[(size_t)(k + 1) * rs + (size_t)n * cs];
  out[t] = pk.u;
}

__global__ void ccnorm_kernel(const float* __restrict__ cb, int C,
                              float* __restrict__ out) {
  int i = blockIdx.x * blockDim.x + threadIdx.x;
  if (i >= C) return;
  float s = 0.f;
  #pragma unroll 4
  for (int d = 0; d < 128; ++d) { float v = cb[i * 128 + d]; s += v * v; }
  out[i] = s;
}

// ---------------------------------------------------------------------------
// Main fused kernel
// ---------------------------------------------------------------------------
struct Params {
  const float* x;
  const float* b1[3];
  const float* b2[3];
  const uint32_t* w1p[3];
  const uint32_t* w2p[3];
  const uint32_t* cbd[3];   // codebook^T fragments (K=128, N=C)  for distances
  const uint32_t* cbq[3];   // codebook fragments   (K=C,  N=128) for weights@c
  const float* ccn[3];
  const uint32_t* f1p;      // fus_w1 fragments (K=384, N=128)
  const uint32_t* f2p;      // fus_w2 fragments (K=128, N=128)
  const float* fb1;
  const float* fb2;
  float* out;
};

__device__ __forceinline__ v8f wmma16(v16h a, v16h b, v8f c) {
  return __builtin_amdgcn_wmma_f32_16x16x32_f16(false, a, false, b,
                                                (short)0, c, false, false);
}

// A fragment (16x32 f16) from a row-major LDS tile, stride in halves.
// Lane holds row M = lane&15; K-halves per ISA layout -> two b128 LDS reads.
__device__ __forceinline__ v16h load_a(const _Float16* buf, int stride,
                                       int kt, int lane) {
  int m = lane & 15;
  int kbase = kt * 32 + ((lane >> 4) << 3);
  const _Float16* r = buf + m * stride + kbase;
  union { uint4 q[2]; v16h v; } u;
  u.q[0] = *(const uint4*)(r);
  u.q[1] = *(const uint4*)(r + 16);
  return u.v;
}

// B fragment from pre-packed global buffer (32 contiguous bytes per lane).
__device__ __forceinline__ v16h load_b(const uint32_t* frag, int lane) {
  const uint4* p = (const uint4*)(frag + lane * 8);
  union { uint4 q[2]; v16h v; } u;
  u.q[0] = p[0];
  u.q[1] = p[1];
  return u.v;
}

// Store accumulator (16x16 f32, C/D layout) to LDS as f16, optional ReLU.
__device__ __forceinline__ void store_d(_Float16* buf, int stride, int ncol0,
                                        int lane, v8f acc, bool relu) {
  int n = ncol0 + (lane & 15);
  int mb = (lane >> 4) << 3;
  #pragma unroll
  for (int r = 0; r < 8; ++r) {
    float v = acc[r];
    if (relu) v = fmaxf(v, 0.f);
    buf[(mb + r) * stride + n] = (_Float16)v;
  }
}

__global__ __launch_bounds__(32 * WAVES)
void hfe_main(Params p) {
  __shared__ alignas(16) _Float16 s_x[WAVES][16][128];
  __shared__ alignas(16) _Float16 s_h[WAVES][16][128];
  __shared__ alignas(16) _Float16 s_q[WAVES][16][384];
  __shared__ alignas(16) _Float16 s_w[WAVES][16][256];
  __shared__ float s_r[WAVES][48];   // [0..15] = row norms, [16..47] scratch

  const int w    = threadIdx.x >> 5;
  const int lane = threadIdx.x & 31;
  const int m16  = lane & 15;
  const int hf   = lane >> 4;
  const long row0 = (long)(blockIdx.x * WAVES + w) * 16;

  _Float16* xb = &s_x[w][0][0];
  _Float16* hb = &s_h[w][0][0];
  _Float16* qb = &s_q[w][0][0];
  _Float16* wb = &s_w[w][0][0];
  float*    rb = s_r[w];

  // ---- load x tile (16x128 f32 -> f16 LDS), fully coalesced b128 loads ----
  {
    const float4* xg = (const float4*)(p.x + row0 * 128);
    for (int i = lane; i < 512; i += 32) {
      float4 f = xg[i];
      _Float16* d = xb + i * 4;
      d[0] = (_Float16)f.x; d[1] = (_Float16)f.y;
      d[2] = (_Float16)f.z; d[3] = (_Float16)f.w;
    }
  }
  __syncthreads();

  const int Cs[3] = {256, 128, 64};

  for (int lvl = 0; lvl < 3; ++lvl) {
    const int C = Cs[lvl];

    // ---- h1 = relu(x @ W1 + b1) -> wb[:, :128] ----
    for (int nt = 0; nt < 8; ++nt) {
      float bv = p.b1[lvl][nt * 16 + m16];
      v8f acc;
      #pragma unroll
      for (int r = 0; r < 8; ++r) acc[r] = bv;
      #pragma unroll
      for (int kt = 0; kt < 4; ++kt) {
        v16h a = load_a(xb, 128, kt, lane);
        v16h b = load_b(p.w1p[lvl] + (kt * 8 + nt) * 256, lane);
        acc = wmma16(a, b, acc);
      }
      store_d(wb, 256, nt * 16, lane, acc, true);
    }
    __syncthreads();

    // ---- h = h1 @ W2 + b2 -> hb ----
    for (int nt = 0; nt < 8; ++nt) {
      float bv = p.b2[lvl][nt * 16 + m16];
      v8f acc;
      #pragma unroll
      for (int r = 0; r < 8; ++r) acc[r] = bv;
      #pragma unroll
      for (int kt = 0; kt < 4; ++kt) {
        v16h a = load_a(wb, 256, kt, lane);
        v16h b = load_b(p.w2p[lvl] + (kt * 8 + nt) * 256, lane);
        acc = wmma16(a, b, acc);
      }
      store_d(hb, 128, nt * 16, lane, acc, false);
    }
    __syncthreads();

    // ---- row norms ||h||^2 (2 lanes per row) ----
    {
      float s = 0.f;
      const _Float16* rr = hb + m16 * 128 + hf * 64;
      #pragma unroll 4
      for (int c = 0; c < 64; ++c) { float v = (float)rr[c]; s += v * v; }
      rb[16 + lane] = s;
      __syncthreads();
      if (lane < 16) rb[lane] = rb[16 + lane] + rb[32 + lane];
      __syncthreads();
    }

    // ---- distances: G = h @ c^T ; d = sqrt(max(hh + cc - 2G, 0)) -> wb ----
    const int NTC = C / 16;
    for (int nt = 0; nt < NTC; ++nt) {
      v8f acc;
      #pragma unroll
      for (int r = 0; r < 8; ++r) acc[r] = 0.f;
      #pragma unroll
      for (int kt = 0; kt < 4; ++kt) {
        v16h a = load_a(hb, 128, kt, lane);
        v16h b = load_b(p.cbd[lvl] + (kt * NTC + nt) * 256, lane);
        acc = wmma16(a, b, acc);
      }
      float cc = p.ccn[lvl][nt * 16 + m16];
      #pragma unroll
      for (int r = 0; r < 8; ++r) {
        int mm = r + hf * 8;
        float d2 = rb[mm] + cc - 2.f * acc[r];
        float d = sqrtf(fmaxf(d2, 0.f));
        wb[mm * 256 + nt * 16 + m16] = (_Float16)d;
      }
    }
    __syncthreads();

    // ---- softmax(-d) per row -> weights (f16) in wb ----
    {
      const int C2 = C >> 1;
      const _Float16* rw = wb + m16 * 256 + hf * C2;
      float mn = 3.0e38f;
      for (int c = 0; c < C2; ++c) mn = fminf(mn, (float)rw[c]);
      rb[16 + lane] = mn;
      __syncthreads();
      mn = fminf(rb[16 + m16], rb[32 + m16]);
      __syncthreads();
      float sum = 0.f;
      for (int c = 0; c < C2; ++c) sum += __expf(mn - (float)rw[c]);
      rb[16 + lane] = sum;
      __syncthreads();
      sum = rb[16 + m16] + rb[32 + m16];
      float inv = 1.f / sum;
      _Float16* ww = wb + m16 * 256 + hf * C2;
      for (int c = 0; c < C2; ++c)
        ww[c] = (_Float16)(__expf(mn - (float)ww[c]) * inv);
      __syncthreads();
    }

    // ---- quantized = weights @ c -> qb[:, lvl*128 : lvl*128+128] ----
    const int KTq = C / 32;
    for (int nt = 0; nt < 8; ++nt) {
      v8f acc;
      #pragma unroll
      for (int r = 0; r < 8; ++r) acc[r] = 0.f;
      for (int kt = 0; kt < KTq; ++kt) {
        v16h a = load_a(wb, 256, kt, lane);
        v16h b = load_b(p.cbq[lvl] + (kt * 8 + nt) * 256, lane);
        acc = wmma16(a, b, acc);
      }
      #pragma unroll
      for (int r = 0; r < 8; ++r) {
        int mm = r + hf * 8;
        qb[mm * 384 + lvl * 128 + nt * 16 + m16] = (_Float16)acc[r];
      }
    }
    __syncthreads();

    // ---- x = h + quantized ----
    for (int t = lane; t < 2048; t += 32) {
      int mm = t >> 7, c = t & 127;
      xb[mm * 128 + c] =
          (_Float16)((float)hb[mm * 128 + c] + (float)qb[mm * 384 + lvl * 128 + c]);
    }
    __syncthreads();
  }

  // ---- fusion: t = relu(concat(q) @ fus_w1 + fb1) -> wb ----
  for (int nt = 0; nt < 8; ++nt) {
    float bv = p.fb1[nt * 16 + m16];
    v8f acc;
    #pragma unroll
    for (int r = 0; r < 8; ++r) acc[r] = bv;
    #pragma unroll
    for (int kt = 0; kt < 12; ++kt) {
      v16h a = load_a(qb, 384, kt, lane);
      v16h b = load_b(p.f1p + (kt * 8 + nt) * 256, lane);
      acc = wmma16(a, b, acc);
    }
    store_d(wb, 256, nt * 16, lane, acc, true);
  }
  __syncthreads();

  // ---- out = t @ fus_w2 + fb2 + x -> global f32 ----
  for (int nt = 0; nt < 8; ++nt) {
    float bv = p.fb2[nt * 16 + m16];
    v8f acc;
    #pragma unroll
    for (int r = 0; r < 8; ++r) acc[r] = bv;
    #pragma unroll
    for (int kt = 0; kt < 4; ++kt) {
      v16h a = load_a(wb, 256, kt, lane);
      v16h b = load_b(p.f2p + (kt * 8 + nt) * 256, lane);
      acc = wmma16(a, b, acc);
    }
    #pragma unroll
    for (int r = 0; r < 8; ++r) {
      int mm = r + hf * 8;
      int col = nt * 16 + m16;
      p.out[(row0 + mm) * 128 + col] = acc[r] + (float)xb[mm * 128 + col];
    }
  }
}

// ---------------------------------------------------------------------------
extern "C" void kernel_launch(void* const* d_in, const int* in_sizes, int n_in,
                              void* d_out, int out_size, void* d_ws, size_t ws_size,
                              hipStream_t stream) {
  const float* x = (const float*)d_in[0];
  const float* ew1[3] = {(const float*)d_in[1], (const float*)d_in[2], (const float*)d_in[3]};
  const float* eb1[3] = {(const float*)d_in[4], (const float*)d_in[5], (const float*)d_in[6]};
  const float* ew2[3] = {(const float*)d_in[7], (const float*)d_in[8], (const float*)d_in[9]};
  const float* eb2[3] = {(const float*)d_in[10], (const float*)d_in[11], (const float*)d_in[12]};
  const float* cb[3]  = {(const float*)d_in[13], (const float*)d_in[14], (const float*)d_in[15]};
  const float* fw1 = (const float*)d_in[16];
  const float* fb1 = (const float*)d_in[17];
  const float* fw2 = (const float*)d_in[18];
  const float* fb2 = (const float*)d_in[19];

  const int Cs[3] = {256, 128, 64};
  uint32_t* ws = (uint32_t*)d_ws;
  size_t off = 0;
  auto alloc = [&](size_t n) { uint32_t* p = ws + off; off += n; return p; };

  uint32_t *w1p[3], *w2p[3], *cbdp[3], *cbqp[3];
  for (int i = 0; i < 3; ++i) w1p[i] = alloc(8192);
  for (int i = 0; i < 3; ++i) w2p[i] = alloc(8192);
  for (int i = 0; i < 3; ++i) cbdp[i] = alloc((size_t)4 * (Cs[i] / 16) * 256);
  for (int i = 0; i < 3; ++i) cbqp[i] = alloc((size_t)(Cs[i] / 32) * 8 * 256);
  uint32_t* f1p = alloc(24576);
  uint32_t* f2p = alloc(8192);
  float* ccn[3];
  for (int i = 0; i < 3; ++i) ccn[i] = (float*)alloc(Cs[i]);

  auto pack = [&](const float* W, int rs, int cs, int K, int N, uint32_t* out) {
    int total = (K / 32) * (N / 16) * 256;
    pack_b_kernel<<<(total + 255) / 256, 256, 0, stream>>>(W, rs, cs, K / 32, N / 16, out);
  };
  for (int i = 0; i < 3; ++i) pack(ew1[i], 128, 1, 128, 128, w1p[i]);
  for (int i = 0; i < 3; ++i) pack(ew2[i], 128, 1, 128, 128, w2p[i]);
  for (int i = 0; i < 3; ++i) pack(cb[i], 1, 128, 128, Cs[i], cbdp[i]);  // c^T
  for (int i = 0; i < 3; ++i) pack(cb[i], 128, 1, Cs[i], 128, cbqp[i]);  // c
  pack(fw1, 128, 1, 384, 128, f1p);
  pack(fw2, 128, 1, 128, 128, f2p);
  for (int i = 0; i < 3; ++i)
    ccnorm_kernel<<<(Cs[i] + 63) / 64, 64, 0, stream>>>(cb[i], Cs[i], ccn[i]);

  Params prm;
  prm.x = x;
  for (int i = 0; i < 3; ++i) {
    prm.b1[i] = eb1[i]; prm.b2[i] = eb2[i];
    prm.w1p[i] = w1p[i]; prm.w2p[i] = w2p[i];
    prm.cbd[i] = cbdp[i]; prm.cbq[i] = cbqp[i];
    prm.ccn[i] = ccn[i];
  }
  prm.f1p = f1p; prm.f2p = f2p; prm.fb1 = fb1; prm.fb2 = fb2;
  prm.out = (float*)d_out;

  int nrows = in_sizes[0] / 128;
  int blocks = nrows / (16 * WAVES);
  hfe_main<<<blocks, 32 * WAVES, 0, stream>>>(prm);
}
